// GraphMMoE_20727512171063
// MI455X (gfx1250) — matrix operand order repaired and checked by reading, not compiled, not run on hardware
//
#include <hip/hip_runtime.h>
#include <hip/hip_bf16.h>
#include <stdint.h>

typedef __attribute__((ext_vector_type(16))) __bf16 v16bf;
typedef __attribute__((ext_vector_type(8)))  __bf16 v8bf;
typedef __attribute__((ext_vector_type(4)))  __bf16 v4bf;
typedef __attribute__((ext_vector_type(8)))  float  v8f;
typedef __attribute__((ext_vector_type(4)))  int    v4i;

#define AS_GLOBAL __attribute__((address_space(1)))
#define AS_LDS    __attribute__((address_space(3)))

#define NB    4096      // batches
#define MM    8         // modalities per batch
#define DD    256       // embed dim
#define HH    256       // hidden dim
#define EE    8         // experts
#define TT    5         // tasks
#define NN    (NB*MM)   // tokens = 32768
#define XST   264       // padded LDS row stride (bf16 elems): row*264*2B -> 4-bank rotation

// Async global->LDS staging (CDNA5), guarded so compile can never fail on it.
#if defined(__gfx1250__) && __has_builtin(__builtin_amdgcn_global_load_async_to_lds_b128) && __has_builtin(__builtin_amdgcn_s_wait_asynccnt)
#define USE_ASYNC_LDS 1
#else
#define USE_ASYNC_LDS 0
#endif

// ---------------------------------------------------------------------------
// Kernel A: gating / losses (tiny: T=5, E=8)
// ---------------------------------------------------------------------------
__device__ inline float cv_ddof1_8(const float* v) {
    float m = 0.f;
    for (int i = 0; i < EE; ++i) m += v[i];
    m *= (1.f / EE);
    float var = 0.f;
    for (int i = 0; i < EE; ++i) { float d = v[i] - m; var += d * d; }
    var *= (1.f / (EE - 1));
    return var / (m * m + 1e-10f);
}

__global__ void gating_kernel(const float* __restrict__ A_logits,
                              const int* __restrict__ task_p,
                              const int* __restrict__ k_p,
                              float* __restrict__ scal,
                              float* __restrict__ wt) {
    if (threadIdx.x != 0 || blockIdx.x != 0) return;
    float w[TT][EE];
    for (int t = 0; t < TT; ++t) {
        float mx = -1e30f;
        for (int e = 0; e < EE; ++e) mx = fmaxf(mx, A_logits[t * EE + e]);
        float s = 0.f;
        for (int e = 0; e < EE; ++e) { w[t][e] = __expf(A_logits[t * EE + e] - mx); s += w[t][e]; }
        float inv = 1.f / s;
        for (int e = 0; e < EE; ++e) w[t][e] *= inv;
    }
    int k = *k_p; if (k < 1) k = 1; if (k > EE) k = EE;
    float mmoe = 0.f;
    for (int t = 0; t < TT; ++t) {
        int idx[EE]; for (int i = 0; i < EE; ++i) idx[i] = i;
        for (int i = 0; i < EE; ++i) {             // selection sort, descending (top_k order)
            int best = i;
            for (int j = i + 1; j < EE; ++j)
                if (w[t][idx[j]] > w[t][idx[best]]) best = j;
            int tmp = idx[i]; idx[i] = idx[best]; idx[best] = tmp;
        }
        float g[EE], gs = 0.f;
        float gm = w[t][idx[0]];
        for (int j = 0; j < k; ++j) { g[j] = __expf(w[t][idx[j]] - gm); gs += g[j]; }
        for (int j = 0; j < k; ++j) g[j] /= gs;
        float imp[EE], ld[EE];
        for (int i = 0; i < EE; ++i) { imp[i] = 0.f; ld[i] = 0.f; }
        for (int j = 0; j < k; ++j) { imp[idx[j]] = (float)NN * g[j]; ld[idx[j]] = (float)NN; }
        mmoe += cv_ddof1_8(imp) + cv_ddof1_8(ld);
    }
    mmoe *= 0.01f;
    // graph loss: A_agg[t] = mean_e w[t][e]
    float l1 = 0.f, usage = 0.f;
    for (int t = 0; t < TT; ++t) {
        float agg = 0.f;
        for (int e = 0; e < EE; ++e) agg += w[t][e];
        agg *= (1.f / EE);
        l1 += fabsf(agg); usage += agg;
    }
    float up = usage / (usage + 1e-10f);
    float entropy = -(up * __logf(up + 1e-10f));
    scal[0] = mmoe;
    scal[1] = l1 - entropy;
    int ti = *task_p; if (ti < 0) ti = 0; if (ti >= TT) ti = TT - 1;
    for (int e = 0; e < EE; ++e) wt[e] = w[ti][e];
}

// ---------------------------------------------------------------------------
// Kernel B1: x fp32 -> bf16
// ---------------------------------------------------------------------------
__global__ void cvt_x_kernel(const float* __restrict__ x, __bf16* __restrict__ xb) {
    int i = blockIdx.x * 256 + threadIdx.x;      // covers NN*DD/4 elements
    float4 v = ((const float4*)x)[i];
    v4bf o = { (__bf16)v.x, (__bf16)v.y, (__bf16)v.z, (__bf16)v.w };
    ((v4bf*)xb)[i] = o;
}

// ---------------------------------------------------------------------------
// Kernel B2: build column-major bf16 weights; fold w[task] into W2
// W1t[e][h][d] = W1[e][d][h]         (B-matrix for stage 1, K=d)
// W2t[e][d][h] = wt[e]*W2[e][h][d]   (B-matrix for stage 2, K=h)
// ---------------------------------------------------------------------------
__global__ void prep_weights_kernel(const float* __restrict__ W1,
                                    const float* __restrict__ W2,
                                    const float* __restrict__ wt,
                                    __bf16* __restrict__ W1t,
                                    __bf16* __restrict__ W2t) {
    int gid = blockIdx.x * 256 + threadIdx.x;    // EE*HH*DD = 524288
    int e = gid >> 16;
    int r = gid & 65535;
    int a = r >> 8;          // write-side major index
    int b = r & 255;         // write-side fastest index (coalesced)
    float src = W1[(e << 16) + (b << 8) + a];
    W1t[gid] = (__bf16)src;
    float src2 = W2[(e << 16) + (b << 8) + a];
    W2t[gid] = (__bf16)(wt[e] * src2);
}

__global__ void b2comb_kernel(const float* __restrict__ b2,
                              const float* __restrict__ wt,
                              float* __restrict__ out) {
    int d = threadIdx.x;
    float s = 0.f;
    for (int e = 0; e < EE; ++e) s += wt[e] * b2[e * DD + d];
    out[d] = s;
}

// ---------------------------------------------------------------------------
// Main fused MoE kernel: per block 64 N-rows x 256 cols, both GEMMs fused.
// ---------------------------------------------------------------------------
__device__ inline v16bf load_a_frag(const __bf16* base) {
    // A-frag (16x32 bf16): lane(row=l%16, half=l/16) holds K = k0+16h+{0..15}
    v8bf lo = *(const v8bf*)(base);
    v8bf hi = *(const v8bf*)(base + 8);
    return __builtin_shufflevector(lo, hi, 0, 1, 2, 3, 4, 5, 6, 7,
                                   8, 9, 10, 11, 12, 13, 14, 15);
}

__global__ __launch_bounds__(256) void moe_gemm_kernel(
    const __bf16* __restrict__ Xg,   // [NN][DD] bf16
    const __bf16* __restrict__ W1t,  // [EE][HH][DD] bf16 (col-major for stage1)
    const __bf16* __restrict__ W2t,  // [EE][DD][HH] bf16, pre-scaled by w[task]
    const float* __restrict__ b1g,   // [EE][HH] f32
    float* __restrict__ mm_ws)       // [NB][DD] f32
{
    __shared__ __align__(16) __bf16 Xs[64 * XST];
    __shared__ __align__(16) __bf16 Hs[64 * XST];

    const int tid  = threadIdx.x;
    const int lane = tid & 31;
    const int wv   = tid >> 5;
    const int rt   = wv & 3;        // row tile (16 rows each)
    const int cH   = wv >> 2;       // column half (128 cols each)
    const int l16  = lane & 15;
    const int half = lane >> 4;
    const int blk  = blockIdx.x;
    const long rowbase = (long)blk * 64;

    // Stage X tile (64x256 bf16) into LDS (padded stride).
#if USE_ASYNC_LDS
    for (int i = tid; i < 64 * 32; i += 256) {
        int r = i >> 5, c8 = (i & 31) << 3;
        // AS1: generic & global addresses are numerically identical (inttoptr).
        // AS3: low 32 bits of an LDS-aperture flat address are the LDS offset (ISA 10.2).
        __builtin_amdgcn_global_load_async_to_lds_b128(
            (AS_GLOBAL v4i*)(uintptr_t)&Xg[(rowbase + r) * DD + c8],
            (AS_LDS v4i*)(uint32_t)(uintptr_t)&Xs[r * XST + c8],
            0, 0);
    }
    __builtin_amdgcn_s_wait_asynccnt(0);
#else
    for (int i = tid; i < 64 * 32; i += 256) {
        int r = i >> 5, c8 = (i & 31) << 3;
        *(v8bf*)(&Xs[r * XST + c8]) = *(const v8bf*)(&Xg[(rowbase + r) * DD + c8]);
    }
#endif
    __syncthreads();

    const v8f vzero = {0.f, 0.f, 0.f, 0.f, 0.f, 0.f, 0.f, 0.f};
    v8f accY[8];
#pragma unroll
    for (int t = 0; t < 8; ++t) accY[t] = vzero;

    const int rowA = rt * 16 + l16;
    const __bf16* Arow = &Xs[rowA * XST];
    const __bf16* Hrow = &Hs[rowA * XST];

    for (int e = 0; e < EE; ++e) {
        const __bf16* W1e = W1t + e * 65536;
        const __bf16* W2e = W2t + e * 65536;
        if (e + 1 < EE) {   // hint next expert's weights into cache
            __builtin_prefetch(W1t + (e + 1) * 65536 + tid * 64, 0, 1);
            __builtin_prefetch(W2t + (e + 1) * 65536 + tid * 64, 0, 1);
        }

        // ---- Stage 1: H = relu(X * W1_e + b1_e) ----
        v8f accH[8];
#pragma unroll
        for (int t = 0; t < 8; ++t) accH[t] = vzero;

        for (int kt = 0; kt < 8; ++kt) {
            const int ka = kt * 32 + 16 * half;
            v16bf a = load_a_frag(Arow + ka);
            const int kb = kt * 32 + 16 * half;
#pragma unroll
            for (int t = 0; t < 8; ++t) {
                const int col = cH * 128 + t * 16 + l16;
                v16bf b = *(const v16bf*)(&W1e[col * DD + kb]);
                accH[t] = __builtin_amdgcn_wmma_f32_16x16x32_bf16(
                    false, a, false, b, (short)0, accH[t], false, false);
            }
        }
        // bias + relu -> bf16 H tile in LDS
#pragma unroll
        for (int t = 0; t < 8; ++t) {
            const int col = cH * 128 + t * 16 + l16;
            const float b1v = b1g[e * HH + col];
#pragma unroll
            for (int i = 0; i < 8; ++i) {
                float h = fmaxf(accH[t][i] + b1v, 0.f);
                Hs[(rt * 16 + 8 * half + i) * XST + col] = (__bf16)h;
            }
        }
        __syncthreads();

        // ---- Stage 2: Y += H * (w_e * W2_e) ----
        for (int kt = 0; kt < 8; ++kt) {
            const int ka = kt * 32 + 16 * half;
            v16bf a = load_a_frag(Hrow + ka);
            const int kb = kt * 32 + 16 * half;
#pragma unroll
            for (int t = 0; t < 8; ++t) {
                const int col = cH * 128 + t * 16 + l16;
                v16bf b = *(const v16bf*)(&W2e[col * HH + kb]);
                accY[t] = __builtin_amdgcn_wmma_f32_16x16x32_bf16(
                    false, a, false, b, (short)0, accY[t], false, false);
            }
        }
        __syncthreads();   // Hs reused next expert
    }

    // Rows i=0..7 of each C tile are exactly m=0..7 of batch (blk*8 + 2*rt + half):
    // mm_moe reduction over M is free here.
    const int batch = blk * 8 + 2 * rt + half;
#pragma unroll
    for (int t = 0; t < 8; ++t) {
        const int col = cH * 128 + t * 16 + l16;
        float s = 0.f;
#pragma unroll
        for (int i = 0; i < 8; ++i) s += accY[t][i];
        mm_ws[batch * DD + col] = s;
    }
}

// ---------------------------------------------------------------------------
// Kernel D: +8*b2comb, LayerNorm(D), head, sigmoid, per-row BCE term
// ---------------------------------------------------------------------------
__global__ __launch_bounds__(256) void ln_head_kernel(
    const float* __restrict__ mm, const float* __restrict__ b2c,
    const float* __restrict__ gamma, const float* __restrict__ beta,
    const float* __restrict__ Wm, const float* __restrict__ bm,
    const float* __restrict__ ty, float* __restrict__ out,
    float* __restrict__ bce_arr)
{
    __shared__ float r1[256];
    __shared__ float r2[256];
    const int b = blockIdx.x, t = threadIdx.x;
    float x = mm[b * DD + t] + 8.f * b2c[t];
    r1[t] = x; r2[t] = x * x;
    __syncthreads();
    for (int s = 128; s > 0; s >>= 1) {
        if (t < s) { r1[t] += r1[t + s]; r2[t] += r2[t + s]; }
        __syncthreads();
    }
    const float mu  = r1[0] * (1.f / DD);
    const float var = r2[0] * (1.f / DD) - mu * mu;
    const float f = gamma[t] * (x - mu) * rsqrtf(var + 1e-5f) + beta[t];
    __syncthreads();
    r1[t] = f * Wm[t];
    __syncthreads();
    for (int s = 128; s > 0; s >>= 1) {
        if (t < s) r1[t] += r1[t + s];
        __syncthreads();
    }
    if (t == 0) {
        float o = r1[0] + bm[0];
        float sg = 1.f / (1.f + __expf(-o));
        out[b] = sg;
        float p = fminf(fmaxf(sg, 1e-7f), 1.f - 1e-7f);
        float y = ty[b];
        bce_arr[b] = -(y * __logf(p) + (1.f - y) * __logf(1.f - p));
    }
}

// ---------------------------------------------------------------------------
// Kernel E: deterministic BCE mean + scalar outputs
// ---------------------------------------------------------------------------
__global__ void finalize_kernel(const float* __restrict__ scal,
                                const float* __restrict__ bce_arr,
                                float* __restrict__ out) {
    __shared__ float r[256];
    int t = threadIdx.x;
    float s = 0.f;
    for (int i = t; i < NB; i += 256) s += bce_arr[i];
    r[t] = s;
    __syncthreads();
    for (int k = 128; k > 0; k >>= 1) {
        if (t < k) r[t] += r[t + k];
        __syncthreads();
    }
    if (t == 0) {
        out[NB + 0] = scal[0];                  // mmoe_total_loss
        out[NB + 1] = r[0] * (1.f / NB);        // pred_loss
        out[NB + 2] = scal[1];                  // graph_loss
    }
}

// ---------------------------------------------------------------------------
extern "C" void kernel_launch(void* const* d_in, const int* in_sizes, int n_in,
                              void* d_out, int out_size, void* d_ws, size_t ws_size,
                              hipStream_t stream) {
    (void)in_sizes; (void)n_in; (void)out_size; (void)ws_size;
    const float* mm_embed = (const float*)d_in[0];
    const float* true_y   = (const float*)d_in[1];
    const float* A_logits = (const float*)d_in[2];
    const float* W1       = (const float*)d_in[3];
    const float* b1       = (const float*)d_in[4];
    const float* W2       = (const float*)d_in[5];
    const float* b2       = (const float*)d_in[6];
    const float* gamma    = (const float*)d_in[7];
    const float* beta     = (const float*)d_in[8];
    const float* Wm       = (const float*)d_in[9];
    const float* bm       = (const float*)d_in[10];
    const int*   task_p   = (const int*)d_in[11];
    const int*   k_p      = (const int*)d_in[12];
    float* out = (float*)d_out;

    char* ws = (char*)d_ws;
    float*  scal = (float*)ws;                 // [0]=mmoe, [1]=graph
    float*  wt   = scal + 16;                  // 8 floats, within first 256B
    size_t off = 256;
    __bf16* Xbf = (__bf16*)(ws + off); off += (size_t)NN * DD * 2;        // 16 MB
    __bf16* W1t = (__bf16*)(ws + off); off += (size_t)EE * HH * DD * 2;   // 1 MB
    __bf16* W2t = (__bf16*)(ws + off); off += (size_t)EE * HH * DD * 2;   // 1 MB
    float*  b2c = (float*)(ws + off);  off += 256 * 4;
    float*  mmw = (float*)(ws + off);  off += (size_t)NB * DD * 4;        // 4 MB
    float*  bce = (float*)(ws + off);  off += (size_t)NB * 4;

    gating_kernel<<<1, 32, 0, stream>>>(A_logits, task_p, k_p, scal, wt);
    cvt_x_kernel<<<(NN * DD / 4) / 256, 256, 0, stream>>>(mm_embed, Xbf);
    prep_weights_kernel<<<(EE * HH * DD) / 256, 256, 0, stream>>>(W1, W2, wt, W1t, W2t);
    b2comb_kernel<<<1, 256, 0, stream>>>(b2, wt, b2c);
    moe_gemm_kernel<<<NN / 64, 256, 0, stream>>>(Xbf, W1t, W2t, b1, mmw);
    ln_head_kernel<<<NB, 256, 0, stream>>>(mmw, b2c, gamma, beta, Wm, bm, true_y, out, bce);
    finalize_kernel<<<1, 256, 0, stream>>>(scal, bce, out);
}
